// ShiftedWindowAttention_59940563583307
// MI455X (gfx1250) — matrix operand order repaired and checked by reading, not compiled, not run on hardware
//
#include <hip/hip_runtime.h>
#include <hip/hip_bf16.h>

typedef __attribute__((ext_vector_type(16))) __bf16 v16bf;
typedef __attribute__((ext_vector_type(8)))  float  v8f;

#define HW_    56
#define SS_    3
#define CDIM_  512
#define NWIN_  2048         // 32 images * 64 windows
#define TOK_   100352       // NWIN_ * 49
#define SCALE_ 0.17677669529663689f  // 32^-0.5

__device__ __forceinline__ unsigned short f2bf(float f) {
  unsigned int u = __builtin_bit_cast(unsigned int, f);
  u += 0x7FFFu + ((u >> 16) & 1u);     // round-to-nearest-even
  return (unsigned short)(u >> 16);
}

union AFrag { v16bf v; uint4 q[2]; };
union BFrag { v16bf v; uint4 q[2]; };

// ---- CDNA5 async global->LDS copy (ASYNCcnt-tracked, no VGPR round trip) ----
__device__ __forceinline__ void async_copy16(void* lds_dst, const void* gsrc) {
  unsigned lds = (unsigned)(uintptr_t)lds_dst;   // LDS aperture: addr[31:0] = LDS offset
  asm volatile("global_load_async_to_lds_b128 %0, %1, off"
               :: "v"(lds), "v"(gsrc) : "memory");
}
__device__ __forceinline__ void wait_async() {
  asm volatile("s_wait_asynccnt 0x0" ::: "memory");
}

// ------------------------------------------------ weights -> bf16, pre-transposed to [k][n]
__global__ __launch_bounds__(256) void cvt_weights(
    const float* __restrict__ qkv_w, const float* __restrict__ proj_w,
    unsigned short* __restrict__ qkvw_t, unsigned short* __restrict__ projw_t) {
  int i = blockIdx.x * 256 + threadIdx.x;
  if (i < 1536 * 512) {                 // i = k*1536 + n
    int k = i / 1536, n = i - 1536 * k;
    qkvw_t[i] = f2bf(qkv_w[n * 512 + k]);
  }
  if (i < 512 * 512) {
    int k = i >> 9, n = i & 511;
    projw_t[i] = f2bf(proj_w[n * 512 + k]);
  }
}

// ------------------------------------------------ WMMA operand loaders (A: 2x ds_load_b128, B: strided k gather)
__device__ __forceinline__ AFrag load_afrag(const unsigned short* arow_u16, int kk, int hi) {
  AFrag a;
  const char* base = (const char*)arow_u16;
  a.q[0] = *(const uint4*)(base + kk * 64 + hi * 16);
  a.q[1] = *(const uint4*)(base + kk * 64 + 32 + hi * 16);
  return a;
}
__device__ __forceinline__ BFrag load_bfrag(const unsigned short* col_at_k0, int ld, int hi) {
  BFrag b;
  unsigned short t[16];
  #pragma unroll
  for (int s = 0; s < 16; ++s) {
    int k = (s < 8) ? (8 * hi + s) : (16 + 8 * hi + (s - 8));
    t[s] = col_at_k0[k * ld];
  }
  __builtin_memcpy(&b.q[0], t, 32);
  return b;
}

// ------------------------------------------------ QKV GEMM
// C[100352,1536] = gathered(x) @ qkv_w^T ; scatter q/v -> [win][head][49][32],
// k -> [win][head][32][64] (head-dim-major, token padded to 64) all bf16
__global__ __launch_bounds__(256) void qkv_gemm(
    const float* __restrict__ x, const unsigned short* __restrict__ wT,
    const float* __restrict__ qkv_b,
    unsigned short* __restrict__ qws, unsigned short* __restrict__ kws,
    unsigned short* __restrict__ vws) {
  __shared__ __align__(16) unsigned short As[32][136];    // [m][k-chunk]
  __shared__ __align__(16) unsigned short Wt[128][136];   // [k-chunk][n-tile]  (K-major)

  const int ntile = blockIdx.x;               // 0..11
  const int mbase = blockIdx.y * 32;
  const int tid = threadIdx.x;
  const int lane = tid & 31;
  const int wave = tid >> 5;
  const int lo = lane & 15, hi = lane >> 4;
  const int rw = wave >> 2;                   // row tile 0..1
  const int cp = wave & 3;                    // col pair 0..3
  const int ntbase = ntile * 128;

  // A staging role: gather row with shift + window partition folded in
  const int arr = tid >> 3, aseg = tid & 7;
  int R = mbase + arr;
  int win = R / 49, t = R - 49 * win;
  int bb = win >> 6, wrem = win & 63, wi = wrem >> 3, wj = wrem & 7;
  int ti = t / 7, tj = t - 7 * ti;
  int sh = wi * 7 + ti + SS_; if (sh >= HW_) sh -= HW_;
  int sw = wj * 7 + tj + SS_; if (sw >= HW_) sw -= HW_;
  const float* asrc = x + ((size_t)(bb * HW_ + sh) * HW_ + sw) * CDIM_;

  // W staging role: K-major rows of pre-transposed weights
  const int krow = tid >> 1, kseg = tid & 1;  // 128 k-rows x 2 segs(64 bf16)
  const unsigned short* wsrc = wT + (size_t)krow * 1536 + ntbase + kseg * 64;

  v8f acc0 = {}; v8f acc1 = {};

  for (int kc = 0; kc < 4; ++kc) {
    const int kbase = kc * 128;
    __syncthreads();
    { // stage W: async global->LDS, all 8 chunks in flight
      const uint4* wp = (const uint4*)(wsrc + (size_t)kbase * 1536);
      #pragma unroll
      for (int v4 = 0; v4 < 8; ++v4)
        async_copy16(&Wt[krow][kseg * 64 + v4 * 8], wp + v4);
    }
    { // stage A: f32 -> bf16 (computes, stays on VGPR path)
      const float4* ap = (const float4*)(asrc + kbase + aseg * 16);
      #pragma unroll
      for (int v4 = 0; v4 < 4; ++v4) {
        float4 f = ap[v4];
        int c0 = aseg * 16 + v4 * 4;
        As[arr][c0 + 0] = f2bf(f.x); As[arr][c0 + 1] = f2bf(f.y);
        As[arr][c0 + 2] = f2bf(f.z); As[arr][c0 + 3] = f2bf(f.w);
      }
    }
    wait_async();
    __syncthreads();

    #pragma unroll
    for (int kk = 0; kk < 4; ++kk) {
      AFrag a = load_afrag(As[rw * 16 + lo], kk, hi);
      BFrag b0 = load_bfrag(&Wt[kk * 32][cp * 32 + lo], 136, hi);
      BFrag b1 = load_bfrag(&Wt[kk * 32][cp * 32 + 16 + lo], 136, hi);
      acc0 = __builtin_amdgcn_wmma_f32_16x16x32_bf16(false, a.v, false, b0.v, (short)0, acc0, false, false);
      acc1 = __builtin_amdgcn_wmma_f32_16x16x32_bf16(false, a.v, false, b1.v, (short)0, acc1, false, false);
    }
  }

  #pragma unroll
  for (int tile = 0; tile < 2; ++tile) {
    v8f acc = tile ? acc1 : acc0;
    int col = ntbase + cp * 32 + tile * 16 + lo;     // 0..1535
    float bias = qkv_b[col];
    int which = col >> 9;                            // 0=q 1=k 2=v
    int cc = col & 511, head = cc >> 5, hd = cc & 31;
    float sc = (which == 0) ? SCALE_ : 1.0f;
    #pragma unroll
    for (int r = 0; r < 8; ++r) {
      int grow = mbase + rw * 16 + r + 8 * hi;
      float val = (acc[r] + bias) * sc;
      int w2 = grow / 49, t2 = grow - 49 * w2;
      unsigned short bv = f2bf(val);
      if (which == 1) {       // k: [win][head][hd][tok(pad 64)]
        kws[((size_t)(w2 * 16 + head) * 32 + hd) * 64 + t2] = bv;
      } else {                // q/v: [win][head][tok][hd]
        unsigned short* op = (which == 0) ? qws : vws;
        op[((size_t)(w2 * 16 + head) * 49 + t2) * 32 + hd] = bv;
      }
    }
  }
}

// ------------------------------------------------ attention (one block per window*head)
__global__ __launch_bounds__(128) void attn_kernel(
    const unsigned short* __restrict__ qws, const unsigned short* __restrict__ kws,
    const unsigned short* __restrict__ vws, const float* __restrict__ rel_tab,
    unsigned short* __restrict__ aout) {
  __shared__ __align__(16) unsigned short Qs[64][40];   // [query][d]
  __shared__ __align__(16) unsigned short Kt[32][72];   // [d][key(pad 64)]
  __shared__ __align__(16) unsigned short Vs[64][40];   // [key][d]
  __shared__ float Ss[64][65];
  __shared__ __align__(16) unsigned short Ps[64][72];   // 144B rows, 16B aligned
  __shared__ float tbl[169];
  __shared__ int regio[49];

  const int win = blockIdx.x >> 4;
  const int head = blockIdx.x & 15;
  const int tid = threadIdx.x;
  const int lane = tid & 31;
  const int wv = tid >> 5;            // row tile 0..3
  const int lo = lane & 15, hi = lane >> 4;
  const size_t gq = (size_t)(win * 16 + head) * 49 * 32;
  const size_t gk = (size_t)(win * 16 + head) * 32 * 64;

  // stage q/v (rows 49..63 zero) and k^T via async copies
  #pragma unroll
  for (int it = 0; it < 2; ++it) {
    int task = tid + it * 128;
    {
      int row = task >> 2, seg = task & 3;
      if (row < 49) {
        async_copy16(&Qs[row][seg * 8], ((const uint4*)(qws + gq + row * 32)) + seg);
        async_copy16(&Vs[row][seg * 8], ((const uint4*)(vws + gq + row * 32)) + seg);
      } else {
        uint4 z = {0, 0, 0, 0};
        *(uint4*)&Qs[row][seg * 8] = z;
        *(uint4*)&Vs[row][seg * 8] = z;
      }
    }
    {
      int row = task >> 3, seg = task & 7;   // 32 rows x 8 segs
      async_copy16(&Kt[row][seg * 8], ((const uint4*)(kws + gk + row * 64)) + seg);
    }
  }
  for (int i = tid; i < 169; i += 128) tbl[i] = rel_tab[i * 16 + head];
  if (tid < 49) {                     // Swin shift-mask region codes
    int wrem = win & 63, wi = wrem >> 3, wj = wrem & 7;
    int ti = tid / 7, tj = tid - 7 * ti;
    int h = wi * 7 + ti, w = wj * 7 + tj;
    int rh = (h < 49) ? 0 : (h < 53 ? 1 : 2);
    int rw = (w < 49) ? 0 : (w < 53 ? 1 : 2);
    regio[tid] = rh * 3 + rw;
  }
  wait_async();
  __syncthreads();

  // S = q @ k^T  (K = HD = 32 -> one WMMA per tile), + rel-pos bias + shift mask
  #pragma unroll
  for (int j = 0; j < 4; ++j) {
    AFrag a = load_afrag(Qs[wv * 16 + lo], 0, hi);
    BFrag bf = load_bfrag(&Kt[0][j * 16 + lo], 72, hi);
    v8f acc = {};
    acc = __builtin_amdgcn_wmma_f32_16x16x32_bf16(false, a.v, false, bf.v, (short)0, acc, false, false);
    #pragma unroll
    for (int r = 0; r < 8; ++r) {
      int n = wv * 16 + r + 8 * hi;
      int m = j * 16 + lo;
      float s = acc[r];
      if (n < 49 && m < 49) {
        int i1 = n / 7, j1 = n - 7 * i1;
        int i2 = m / 7, j2 = m - 7 * i2;
        s += tbl[(i1 - i2 + 6) * 13 + (j1 - j2 + 6)];
        if (regio[n] != regio[m]) s -= 100.0f;
      } else {
        s = -1e30f;
      }
      Ss[n][m] = s;
    }
  }
  __syncthreads();

  // row softmax; write P bf16 with cols>=49 forced to exact zero
  if (tid < 64) {
    int row = tid;
    if (row < 49) {
      float mx = -1e30f;
      for (int c = 0; c < 49; ++c) mx = fmaxf(mx, Ss[row][c]);
      float sum = 0.f;
      for (int c = 0; c < 49; ++c) { float e = __expf(Ss[row][c] - mx); Ss[row][c] = e; sum += e; }
      float inv = 1.0f / sum;
      for (int c = 0; c < 49; ++c) Ps[row][c] = f2bf(Ss[row][c] * inv);
      for (int c = 49; c < 64; ++c) Ps[row][c] = 0;
    } else {
      for (int c = 0; c < 64; ++c) Ps[row][c] = 0;
    }
  }
  __syncthreads();

  // out = P @ v  (K = 64 -> 2 WMMA k-steps)
  #pragma unroll
  for (int nj = 0; nj < 2; ++nj) {
    v8f acc = {};
    #pragma unroll
    for (int kk = 0; kk < 2; ++kk) {
      AFrag a = load_afrag(Ps[wv * 16 + lo], kk, hi);
      BFrag bf = load_bfrag(&Vs[kk * 32][nj * 16 + lo], 40, hi);
      acc = __builtin_amdgcn_wmma_f32_16x16x32_bf16(false, a.v, false, bf.v, (short)0, acc, false, false);
    }
    #pragma unroll
    for (int r = 0; r < 8; ++r) {
      int n = wv * 16 + r + 8 * hi;
      if (n < 49)
        aout[((size_t)win * 49 + n) * 512 + head * 32 + nj * 16 + lo] = f2bf(acc[r]);
    }
  }
}

// ------------------------------------------------ proj GEMM + reverse shift scatter
__global__ __launch_bounds__(256) void proj_gemm(
    const unsigned short* __restrict__ aio, const unsigned short* __restrict__ wT,
    const float* __restrict__ proj_b, float* __restrict__ out) {
  __shared__ __align__(16) unsigned short As[32][136];
  __shared__ __align__(16) unsigned short Wt[128][136];   // [k][n] K-major

  const int ntile = blockIdx.x;               // 0..3
  const int mbase = blockIdx.y * 32;
  const int tid = threadIdx.x;
  const int lane = tid & 31;
  const int wave = tid >> 5;
  const int lo = lane & 15, hi = lane >> 4;
  const int rw = wave >> 2, cp = wave & 3;
  const int ntbase = ntile * 128;

  const int arr = tid >> 3, aseg = tid & 7;
  const unsigned short* asrc = aio + (size_t)(mbase + arr) * 512;
  const int krow = tid >> 1, kseg = tid & 1;
  const unsigned short* wsrc = wT + (size_t)krow * 512 + ntbase + kseg * 64;

  v8f acc0 = {}; v8f acc1 = {};

  for (int kc = 0; kc < 4; ++kc) {
    const int kbase = kc * 128;
    __syncthreads();
    { // A tile: pure bf16 copy -> async
      const uint4* ap = (const uint4*)(asrc + kbase + aseg * 16);
      async_copy16(&As[arr][aseg * 16 + 0], ap + 0);
      async_copy16(&As[arr][aseg * 16 + 8], ap + 1);
    }
    { // W tile: async
      const uint4* wp = (const uint4*)(wsrc + (size_t)kbase * 512);
      #pragma unroll
      for (int v4 = 0; v4 < 8; ++v4)
        async_copy16(&Wt[krow][kseg * 64 + v4 * 8], wp + v4);
    }
    wait_async();
    __syncthreads();

    #pragma unroll
    for (int kk = 0; kk < 4; ++kk) {
      AFrag a = load_afrag(As[rw * 16 + lo], kk, hi);
      BFrag b0 = load_bfrag(&Wt[kk * 32][cp * 32 + lo], 136, hi);
      BFrag b1 = load_bfrag(&Wt[kk * 32][cp * 32 + 16 + lo], 136, hi);
      acc0 = __builtin_amdgcn_wmma_f32_16x16x32_bf16(false, a.v, false, b0.v, (short)0, acc0, false, false);
      acc1 = __builtin_amdgcn_wmma_f32_16x16x32_bf16(false, a.v, false, b1.v, (short)0, acc1, false, false);
    }
  }

  #pragma unroll
  for (int tile = 0; tile < 2; ++tile) {
    v8f acc = tile ? acc1 : acc0;
    int col = ntbase + cp * 32 + tile * 16 + lo;
    float bias = proj_b[col];
    #pragma unroll
    for (int r = 0; r < 8; ++r) {
      int grow = mbase + rw * 16 + r + 8 * hi;
      float val = acc[r] + bias;
      int win = grow / 49, t = grow - 49 * win;
      int bb = win >> 6, wrem = win & 63, wi = wrem >> 3, wj = wrem & 7;
      int ti = t / 7, tj = t - 7 * ti;
      int h = wi * 7 + ti + SS_; if (h >= HW_) h -= HW_;
      int w = wj * 7 + tj + SS_; if (w >= HW_) w -= HW_;
      out[((size_t)(bb * HW_ + h) * HW_ + w) * 512 + col] = val;
    }
  }
}

// ------------------------------------------------
extern "C" void kernel_launch(void* const* d_in, const int* in_sizes, int n_in,
                              void* d_out, int out_size, void* d_ws, size_t ws_size,
                              hipStream_t stream) {
  (void)in_sizes; (void)n_in; (void)out_size; (void)ws_size;
  const float* x      = (const float*)d_in[0];
  const float* qkv_w  = (const float*)d_in[1];
  const float* qkv_b  = (const float*)d_in[2];
  const float* proj_w = (const float*)d_in[3];
  const float* proj_b = (const float*)d_in[4];
  const float* rel    = (const float*)d_in[5];
  float* out = (float*)d_out;

  char* wsb = (char*)d_ws;
  size_t off = 0;
  auto take = [&](size_t bytes) -> void* {
    void* p = wsb + off;
    off = (off + bytes + 255) & ~(size_t)255;
    return p;
  };
  unsigned short* qkvw_t = (unsigned short*)take((size_t)1536 * 512 * 2);
  unsigned short* projw_t = (unsigned short*)take((size_t)512 * 512 * 2);
  unsigned short* qws = (unsigned short*)take((size_t)NWIN_ * 16 * 49 * 32 * 2);
  unsigned short* kws = (unsigned short*)take((size_t)NWIN_ * 16 * 32 * 64 * 2);  // padded tok
  unsigned short* vws = (unsigned short*)take((size_t)NWIN_ * 16 * 49 * 32 * 2);
  unsigned short* aout = (unsigned short*)take((size_t)TOK_ * CDIM_ * 2);

  cvt_weights<<<3072, 256, 0, stream>>>(qkv_w, proj_w, qkvw_t, projw_t);
  qkv_gemm<<<dim3(12, TOK_ / 32), 256, 0, stream>>>(x, qkvw_t, qkv_b, qws, kws, vws);
  attn_kernel<<<NWIN_ * 16, 128, 0, stream>>>(qws, kws, vws, rel, aout);
  proj_gemm<<<dim3(4, TOK_ / 32), 256, 0, stream>>>(aout, projw_t, proj_b, out);
}
